// RNNCell_Encoder_2224793059747
// MI455X (gfx1250) — matrix operand order repaired and checked by reading, not compiled, not run on hardware
//
#include <hip/hip_runtime.h>
#include <hip/hip_bf16.h>
#include <math.h>

// ---- problem constants -----------------------------------------------------
#define SEQ     512
#define BATCH   256
#define IN_DIM  300
#define HID     300
#define NP      320   // padded N (HID) -> 20 tiles of 16
#define KP      320   // padded K       -> 10 chunks of 32
#define NT      20    // N tiles
#define KC      10    // K chunks
#define XP_STRIDE 304 // x_proj row stride in floats (16-elem aligned)

// ---- WMMA vector types (CDNA5, wave32) -------------------------------------
typedef __attribute__((ext_vector_type(16))) __bf16         v16bf;
typedef __attribute__((ext_vector_type(8)))  float          v8f;
typedef __attribute__((ext_vector_type(8)))  unsigned int   v8u;

// fp32 -> bf16 via native convert (RNE); returns raw bits for u16 buffers
__device__ __forceinline__ unsigned short f2bf(float f) {
    __bf16 h = (__bf16)f;
    return __builtin_bit_cast(unsigned short, h);
}

// ---------------------------------------------------------------------------
// Pack W_ih / W_hh (both [HID][K], fp32, row = output feature) into bf16
// B-matrix WMMA fragments: frag index = (kc*NT + nt), 32 lanes x 16 bf16.
// Lane l: N = nt*16 + (l&15); value i (0..15): K = kc*32 + (i>>3)*16 + (l>>4)*8 + (i&7).
// Zero-padded to NP x KP.
// ---------------------------------------------------------------------------
__global__ void pack_weights(const float* __restrict__ Wih,
                             const float* __restrict__ Whh,
                             unsigned short* __restrict__ BihP,
                             unsigned short* __restrict__ BhhP) {
    int idx = blockIdx.x * blockDim.x + threadIdx.x;     // ((kc*NT+nt)*32+lane)*16+i
    if (idx >= KC * NT * 32 * 16) return;
    int i    = idx & 15;
    int lane = (idx >> 4) & 31;
    int rest = idx >> 9;            // kc*NT + nt
    int nt   = rest % NT;
    int kc   = rest / NT;
    int n = nt * 16 + (lane & 15);
    int k = kc * 32 + ((i < 8) ? 0 : 16) + ((lane >> 4) << 3) + (i & 7);
    float a = (n < HID && k < IN_DIM) ? Wih[n * IN_DIM + k] : 0.0f;
    float b = (n < HID && k < HID)    ? Whh[n * HID    + k] : 0.0f;
    BihP[idx] = f2bf(a);
    BhhP[idx] = f2bf(b);
}

// zero both padded bf16 hidden-state buffers (h0 = 0, padding stays 0)
__global__ void zero_h(unsigned short* __restrict__ h0,
                       unsigned short* __restrict__ h1) {
    int idx = blockIdx.x * blockDim.x + threadIdx.x;
    if (idx < BATCH * KP) { h0[idx] = 0; h1[idx] = 0; }
}

// ---------------------------------------------------------------------------
// Phase 1: x_proj[row][n] = sum_k x[row][k] * W_ih[n][k] + b_ih[n]
// One wave computes 16 rows x 320 cols (20 N-tiles), reading x exactly once.
// ---------------------------------------------------------------------------
__global__ __launch_bounds__(128)
void xproj_gemm(const float* __restrict__ X,              // [SEQ*BATCH][IN_DIM]
                const unsigned short* __restrict__ BihP,  // packed fragments
                const float* __restrict__ bih,
                float* __restrict__ XP) {                 // [SEQ*BATCH][XP_STRIDE]
    const int lane = threadIdx.x & 31;
    const int wave = threadIdx.x >> 5;
    const int row0 = blockIdx.x * 64 + wave * 16;
    const int hi   = lane >> 4;          // lane half (K-phase select)
    const int lm   = lane & 15;          // A row / C column
    const float* xrow = X + (size_t)(row0 + lm) * IN_DIM;
    const v8u* bbase = (const v8u*)BihP; // 32-byte fragments

    const v8f vzero = {};
    v8f acc[NT];
    #pragma unroll
    for (int t = 0; t < NT; ++t) acc[t] = vzero;

    for (int kc = 0; kc < KC; ++kc) {
        if (kc + 1 < KC)
            __builtin_prefetch(xrow + (kc + 1) * 32 + hi * 8, 0, 0);

        // Build A fragment directly as bf16 (native v_cvt pack, RNE).
        // K pairs are even so a single k0 < IN_DIM test covers both elements.
        v16bf afrag;
        #pragma unroll
        for (int v = 0; v < 8; ++v) {
            int k0 = kc * 32 + ((v < 4) ? 0 : 16) + hi * 8 + 2 * (v & 3);
            float f0 = 0.0f, f1 = 0.0f;
            if (k0 < IN_DIM) { f0 = xrow[k0]; f1 = xrow[k0 + 1]; }
            afrag[2 * v]     = (__bf16)f0;
            afrag[2 * v + 1] = (__bf16)f1;
        }

        #pragma unroll
        for (int t = 0; t < NT; ++t) {
            v8u raw = bbase[(size_t)(kc * NT + t) * 32 + lane];
            v16bf bfrag = __builtin_bit_cast(v16bf, raw);
            acc[t] = __builtin_amdgcn_wmma_f32_16x16x32_bf16(
                false, afrag, false, bfrag, (short)0, acc[t], false, false);
        }
    }

    // Epilogue: C layout -> VGPR r holds (M = r + 8*hi, N = lm) ; add bias
    #pragma unroll
    for (int t = 0; t < NT; ++t) {
        int n = t * 16 + lm;
        if (n < HID) {
            float bias = bih[n];
            #pragma unroll
            for (int r = 0; r < 8; ++r) {
                int m = hi * 8 + r;
                XP[(size_t)(row0 + m) * XP_STRIDE + n] = acc[t][r] + bias;
            }
        }
    }
}

// ---------------------------------------------------------------------------
// Phase 2 (sequential, one launch per timestep, graph-replayed):
//   h_new = tanh(x_proj[s] + h_prev @ W_hh^T + b_hh)
// 80 waves: wave = (m-tile, group of 4 n-tiles). h kept as bf16 [BATCH][KP],
// zero-padded so the K loop needs no guards. On the last step the fp32 result
// is also written to d_out.
// ---------------------------------------------------------------------------
__global__ __launch_bounds__(128)
void rnn_step(const unsigned short* __restrict__ hPrev, // [BATCH][KP] bf16
              unsigned short* __restrict__ hNext,       // [BATCH][KP] bf16
              const unsigned short* __restrict__ BhhP,  // packed fragments
              const float* __restrict__ XP,             // x_proj
              const float* __restrict__ bhh,
              float* __restrict__ out,                  // [BATCH][HID] fp32
              int s, int last) {
    const int lane = threadIdx.x & 31;
    const int wave = threadIdx.x >> 5;
    const int wid  = blockIdx.x * 4 + wave;  // 0..79
    const int mt   = wid / 5;                // 0..15  (batch tile)
    const int ng   = wid % 5;                // 0..4   (n-tile group of 4)
    const int hi   = lane >> 4;
    const int lm   = lane & 15;
    const int row0 = mt * 16;
    const unsigned short* hrow = hPrev + (size_t)(row0 + lm) * KP;
    const v8u* bbase = (const v8u*)BhhP;

    const v8f vzero = {};
    v8f acc[4];
    #pragma unroll
    for (int t = 0; t < 4; ++t) acc[t] = vzero;

    #pragma unroll
    for (int kc = 0; kc < KC; ++kc) {
        // A fragment: packed bf16 pairs gathered as dwords (padded -> no guards)
        v8u au;
        #pragma unroll
        for (int v = 0; v < 8; ++v) {
            int k0 = kc * 32 + ((v < 4) ? 0 : 16) + hi * 8 + 2 * (v & 3);
            au[v] = *(const unsigned int*)(hrow + k0);
        }
        v16bf afrag = __builtin_bit_cast(v16bf, au);

        #pragma unroll
        for (int t = 0; t < 4; ++t) {
            int nt = ng * 4 + t;
            v8u raw = bbase[(size_t)(kc * NT + nt) * 32 + lane];
            v16bf bfrag = __builtin_bit_cast(v16bf, raw);
            acc[t] = __builtin_amdgcn_wmma_f32_16x16x32_bf16(
                false, afrag, false, bfrag, (short)0, acc[t], false, false);
        }
    }

    const float* xp = XP + (size_t)s * BATCH * XP_STRIDE;
    #pragma unroll
    for (int t = 0; t < 4; ++t) {
        int n = (ng * 4 + t) * 16 + lm;
        float bias = (n < HID) ? bhh[n] : 0.0f;
        #pragma unroll
        for (int r = 0; r < 8; ++r) {
            int brow = row0 + hi * 8 + r;
            if (n < HID) {
                float v  = acc[t][r] + xp[(size_t)brow * XP_STRIDE + n] + bias;
                float th = tanhf(v);
                hNext[(size_t)brow * KP + n] = f2bf(th);
                if (last) out[(size_t)brow * HID + n] = th;
            } else {
                hNext[(size_t)brow * KP + n] = 0;  // keep padding zero
            }
        }
    }
}

// ---------------------------------------------------------------------------
extern "C" void kernel_launch(void* const* d_in, const int* in_sizes, int n_in,
                              void* d_out, int out_size, void* d_ws, size_t ws_size,
                              hipStream_t stream) {
    (void)in_sizes; (void)n_in; (void)out_size; (void)ws_size;

    const float* X   = (const float*)d_in[0];  // [SEQ][BATCH][IN_DIM]
    const float* Wih = (const float*)d_in[1];  // [HID][IN_DIM]
    const float* Whh = (const float*)d_in[2];  // [HID][HID]
    const float* bih = (const float*)d_in[3];  // [HID]
    const float* bhh = (const float*)d_in[4];  // [HID]
    float* out = (float*)d_out;                // [BATCH][HID]

    // workspace layout (all offsets 256B-aligned); total ~160.2 MB
    char* ws = (char*)d_ws;
    size_t off = 0;
    float* XP = (float*)(ws + off);            off += (size_t)SEQ * BATCH * XP_STRIDE * 4;
    unsigned short* BihP = (unsigned short*)(ws + off); off += (size_t)KC * NT * 32 * 16 * 2;
    unsigned short* BhhP = (unsigned short*)(ws + off); off += (size_t)KC * NT * 32 * 16 * 2;
    unsigned short* h0   = (unsigned short*)(ws + off); off += (size_t)BATCH * KP * 2;
    unsigned short* h1   = (unsigned short*)(ws + off); off += (size_t)BATCH * KP * 2;

    // phase 0: pack weights into WMMA fragment order + zero hidden state
    pack_weights<<<(KC * NT * 32 * 16 + 255) / 256, 256, 0, stream>>>(Wih, Whh, BihP, BhhP);
    zero_h<<<(BATCH * KP + 255) / 256, 256, 0, stream>>>(h0, h1);

    // phase 1: one streaming GEMM for all input projections
    xproj_gemm<<<SEQ * BATCH / 64, 128, 0, stream>>>(X, BihP, bih, XP);

    // phase 2: sequential scan, double-buffered bf16 state
    unsigned short* hp = h0;
    unsigned short* hn = h1;
    for (int s = 0; s < SEQ; ++s) {
        rnn_step<<<20, 128, 0, stream>>>(hp, hn, BhhP, XP, bhh, out,
                                         s, (s == SEQ - 1) ? 1 : 0);
        unsigned short* tmp = hp; hp = hn; hn = tmp;
    }
}
